// WindowedMultiHeadAttention_58377195487710
// MI455X (gfx1250) — matrix operand order repaired and checked by reading, not compiled, not run on hardware
//
#include <hip/hip_runtime.h>

// ---------------------------------------------------------------------------
// Windowed multi-head attention (Swin) for MI455X / gfx1250 (wave32, WMMA).
//
//   k0: gather relative-position bias    -> ws.bias [H,N,N] fp32
//   k1: qkv = x @ w_qkv + b (bf16 WMMA)  -> ws.q/k/v [B,H,N,D] bf16 (q scaled)
//   k2: fused attn per (window,head)     -> d_out.attn fp32, ws.ctx bf16
//   k3: out = ctx @ w_proj + b (bf16 WMMA, async-LDS A tiles) -> d_out.out
//
// Grid order: column tile fastest so A row-panels stay L2-resident
// (x = 205 MB read ~once instead of 12x; W matrices fully L2-resident).
// ---------------------------------------------------------------------------

typedef __bf16 v16bf __attribute__((ext_vector_type(16)));
typedef __bf16 v8bf  __attribute__((ext_vector_type(8)));
typedef float  v8f   __attribute__((ext_vector_type(8)));

#define TOK 49
#define NH  16
#define CH  512
#define HD  32
#define NWIN 64
#define NB  2048
#define ROWS (NB * TOK)  // 100352 = 784 * 128
__constant__ const float QK_SCALE = 0.17677669529663687f;

__device__ __forceinline__ __bf16 f2bf(float f) {
  unsigned u = __float_as_uint(f);
  u = (u + 0x7FFFu + ((u >> 16) & 1u)) >> 16;  // round-to-nearest-even
  unsigned short s = (unsigned short)u;
  __bf16 r;
  __builtin_memcpy(&r, &s, sizeof(r));
  return r;
}

__device__ __forceinline__ v16bf cat16(v8bf lo, v8bf hi) {
  return __builtin_shufflevector(lo, hi, 0, 1, 2, 3, 4, 5, 6, 7,
                                         8, 9, 10, 11, 12, 13, 14, 15);
}

// A fragment (16xK bf16): lane row M = lane%16; K runs of 8 split by lane-hi.
__device__ __forceinline__ v16bf frag_a(const __bf16* row, int lane, int kc) {
  const int khi = kc + ((lane >> 4) << 3);
  v8bf lo = *(const v8bf*)(row + khi);
  v8bf hi = *(const v8bf*)(row + khi + 16);
  return cat16(lo, hi);
}

// B fragment (Kx16): lane col N = lane%16; K runs of 8 split by lane-hi.
__device__ __forceinline__ v16bf frag_b(const __bf16* row, int lane, int kc) {
  const int k0 = kc + ((lane >> 4) << 3);
  v8bf lo = *(const v8bf*)(row + k0);
  v8bf hi = *(const v8bf*)(row + k0 + 16);
  return cat16(lo, hi);
}

__device__ __forceinline__ v8f wmma_bf16(v16bf a, v16bf b, v8f c) {
  return __builtin_amdgcn_wmma_f32_16x16x32_bf16(false, a, false, b,
                                                 (short)0, c, false, false);
}

// Async global->LDS byte copy (8B per active lane), tracked by ASYNCcnt.
// LDS generic address low 32 bits == LDS byte offset (ISA 10.2 aperture rule).
__device__ __forceinline__ void async_copy_b64(unsigned lds_byte_addr,
                                               unsigned src_byte_off,
                                               const void* sbase) {
  asm volatile("global_load_async_to_lds_b64 %0, %1, %2"
               :
               : "v"(lds_byte_addr), "v"(src_byte_off), "s"(sbase)
               : "memory");
}
__device__ __forceinline__ void wait_async0() {
  asm volatile("s_wait_asynccnt 0x0" ::: "memory");
}

// ---------------------------------------------------------------------------
// k0: bias[h][i][j] = bias_table[rel_index[i*49+j]][h]
// ---------------------------------------------------------------------------
__global__ void bias_gather_kernel(const float* __restrict__ table,
                                   const int* __restrict__ rel,
                                   float* __restrict__ bias_hnn) {
  const int t = blockIdx.x * 256 + threadIdx.x;
  if (t >= NH * TOK * TOK) return;
  const int h = t / (TOK * TOK);
  const int r = t - h * (TOK * TOK);
  bias_hnn[t] = table[rel[r] * NH + h];
}

// ---------------------------------------------------------------------------
// GEMM: [ROWS x 512] @ [512 x LDW], block tile 128x128, 8 waves (4x2),
// wave computes 2x4 16x16 tiles, K-chunks of 32 -> 8 WMMAs per chunk.
// MODE 0: A fp32 (x), LDW=1536, fused b_qkv (+q scale), scatter q/k/v bf16
// MODE 1: A bf16 (ctx, async-LDS staged), LDW=512, fused b_proj, fp32 out
// ---------------------------------------------------------------------------
template <int MODE>
__global__ __launch_bounds__(256) void gemm_kernel(
    const void* __restrict__ Ap, const float* __restrict__ W,
    const float* __restrict__ bvec, __bf16* __restrict__ qo,
    __bf16* __restrict__ ko, __bf16* __restrict__ vo,
    float* __restrict__ fo) {
  constexpr int LDW = (MODE == 0) ? (3 * CH) : CH;
  __shared__ __align__(16) __bf16 As[128][40];   // 40-pad: conflict-free frags
  __shared__ __align__(16) __bf16 Bs[128][40];   // transposed [n][k]

  const int tid  = threadIdx.x;
  const int lane = tid & 31;
  const int wave = tid >> 5;
  const int wr   = wave >> 1;  // 0..3: 32-row strip
  const int wc   = wave & 1;   // 0..1: 64-col strip
  const int col0 = blockIdx.x * 128;  // fastest-varying: share A panel in L2
  const int row0 = blockIdx.y * 128;

  const float*  Af = (const float*)Ap;
  const __bf16* Ab = (const __bf16*)Ap;

  const v8f vz = {0.f, 0.f, 0.f, 0.f, 0.f, 0.f, 0.f, 0.f};
  v8f acc[2][4];
#pragma unroll
  for (int i = 0; i < 2; ++i)
#pragma unroll
    for (int j = 0; j < 4; ++j) acc[i][j] = vz;

  const int ar = tid >> 3;        // A: row 0..31 (stride 32), 4 cols each
  const int ac = (tid & 7) << 2;
  const int bn = tid & 127;       // B: 128 cols, 2 k-groups of 16
  const int bkg = tid >> 7;

  for (int kc = 0; kc < CH; kc += 32) {
    // ---- stage A tile [128 x 32] ----
    if constexpr (MODE == 0) {
#pragma unroll
      for (int rr = 0; rr < 4; ++rr) {
        const int r = ar + (rr << 5);
        const size_t g = (size_t)(row0 + r) * CH + kc + ac;
#pragma unroll
        for (int i = 0; i < 4; ++i) As[r][ac + i] = f2bf(Af[g + i]);
      }
    } else {  // bf16 source: async DMA straight into LDS
#pragma unroll
      for (int rr = 0; rr < 4; ++rr) {
        const int r = ar + (rr << 5);
        const unsigned lds = (unsigned)(uintptr_t)&As[r][ac];
        const unsigned src = (unsigned)(((row0 + r) * CH + kc + ac) * 2);
        async_copy_b64(lds, src, Ab);
      }
    }
    // ---- stage B tile [32 x 128] transposed; coalesced over n per k ----
#pragma unroll
    for (int i = 0; i < 16; ++i) {
      const int k = (bkg << 4) + i;
      Bs[bn][k] = f2bf(W[(size_t)(kc + k) * LDW + col0 + bn]);
    }
    if constexpr (MODE == 0) {
      if (kc + 32 < CH) {  // speculative prefetch (global_prefetch_b8)
        const size_t nx = (size_t)(row0 + ar) * CH + kc + 32 + ac;
        __builtin_prefetch((const char*)Ap + nx * 4, 0, 0);
      }
    } else {
      wait_async0();
    }
    __syncthreads();

    const v16bf a0 = frag_a(&As[(wr << 5) + (lane & 15)][0], lane, 0);
    const v16bf a1 = frag_a(&As[(wr << 5) + 16 + (lane & 15)][0], lane, 0);
#pragma unroll
    for (int j = 0; j < 4; ++j) {
      const v16bf bb = frag_b(&Bs[(wc << 6) + (j << 4) + (lane & 15)][0],
                              lane, 0);
      acc[0][j] = wmma_bf16(a0, bb, acc[0][j]);
      acc[1][j] = wmma_bf16(a1, bb, acc[1][j]);
    }
    __syncthreads();
  }

  // ---- epilogue: row = vr + 8*(lane/16), col = lane%16 ----
#pragma unroll
  for (int i = 0; i < 2; ++i)
#pragma unroll
    for (int j = 0; j < 4; ++j)
#pragma unroll
      for (int vr = 0; vr < 8; ++vr) {
        const int gr = row0 + (wr << 5) + (i << 4) + vr + ((lane >> 4) << 3);
        const int gc = col0 + (wc << 6) + (j << 4) + (lane & 15);
        const float val = acc[i][j][vr] + bvec[gc];
        if constexpr (MODE == 0) {
          const int t = gc >> 9, c = gc & 511;
          const int h = c >> 5, d = c & 31;
          const int bw = gr / TOK, n = gr - bw * TOK;
          const size_t dst = ((size_t)(bw * NH + h) * TOK + n) * HD + d;
          if (t == 0)      qo[dst] = f2bf(val * QK_SCALE);
          else if (t == 1) ko[dst] = f2bf(val);
          else             vo[dst] = f2bf(val);
        } else {
          fo[(size_t)gr * CH + gc] = val;
        }
      }
}

// ---------------------------------------------------------------------------
// k2: fused attention per (window b, head h). 4 waves, 49 padded to 64.
// ---------------------------------------------------------------------------
__global__ __launch_bounds__(128) void attn_kernel(
    const __bf16* __restrict__ q, const __bf16* __restrict__ k,
    const __bf16* __restrict__ v, const float* __restrict__ bias_hnn,
    const float* __restrict__ mask, float* __restrict__ attn_out,
    __bf16* __restrict__ ctx) {
  __shared__ __align__(16) __bf16 qs[64][40];
  __shared__ __align__(16) __bf16 ks[64][40];
  __shared__ __align__(16) __bf16 vt[32][72];  // V transposed [d][token]
  __shared__ __align__(16) __bf16 ps[64][72];  // probs (A operand for P@V)
  __shared__ float ss[64][65];                 // scores (65-pad: no conflicts)

  const int bh   = blockIdx.x;
  const int b    = bh >> 4;
  const int h    = bh & 15;
  const int wm   = b & (NWIN - 1);
  const int tid  = threadIdx.x;
  const int lane = tid & 31;
  const int wave = tid >> 5;
  const __bf16 bz = f2bf(0.f);

  const __bf16* qb = q + (size_t)bh * TOK * HD;
  const __bf16* kb = k + (size_t)bh * TOK * HD;
  const __bf16* vb = v + (size_t)bh * TOK * HD;

  // async DMA the valid 49x32 q/k tiles into padded LDS (8B chunks)
  {
    const unsigned q0 = (unsigned)(uintptr_t)&qs[0][0];
    const unsigned k0 = (unsigned)(uintptr_t)&ks[0][0];
    for (int c = tid; c < (TOK * HD) / 4; c += 128) {  // 392 chunks
      const int e0 = c << 2;                           // bf16 element index
      const int r = e0 >> 5, d = e0 & 31;
      const unsigned off = (unsigned)(r * 80 + d * 2); // row stride 80 B
      const unsigned src = (unsigned)(c << 3);
      async_copy_b64(q0 + off, src, qb);
      async_copy_b64(k0 + off, src, kb);
    }
  }
  // zero pad rows 49..63 of q/k
  for (int idx = tid; idx < 15 * HD; idx += 128) {
    const int r = TOK + (idx >> 5), d = idx & 31;
    qs[r][d] = bz;
    ks[r][d] = bz;
  }
  // V transposed (2B-granular gather: regular loads) + clear probs
  for (int idx = tid; idx < HD * 64; idx += 128) {
    const int d = idx >> 6, t = idx & 63;
    vt[d][t] = (t < TOK) ? vb[t * HD + d] : bz;
  }
  for (int idx = tid; idx < 64 * 72; idx += 128) (&ps[0][0])[idx] = bz;
  // pre-stage bias+mask into score array (branchless add later)
  {
    const float* bh_bias = bias_hnn + h * (TOK * TOK);
    const float* bm = mask + (size_t)wm * (TOK * TOK);
    for (int idx = tid; idx < 64 * 64; idx += 128) {
      const int i = idx >> 6, c = idx & 63;
      ss[i][c] = (i < TOK && c < TOK) ? bh_bias[i * TOK + c] + bm[i * TOK + c]
                                      : 0.f;
    }
  }
  wait_async0();
  __syncthreads();

  // S = q @ k^T : wave owns one 16-row strip; K=32 = one WMMA per tile
  const v8f vz = {0.f, 0.f, 0.f, 0.f, 0.f, 0.f, 0.f, 0.f};
  v8f sacc[4];
  {
    const v16bf a = frag_a(&qs[(wave << 4) + (lane & 15)][0], lane, 0);
#pragma unroll
    for (int j = 0; j < 4; ++j) {
      const v16bf bb = frag_b(&ks[(j << 4) + (lane & 15)][0], lane, 0);
      sacc[j] = wmma_bf16(a, bb, vz);
    }
  }
  // unconditional 1:1 accumulate into pre-staged bias+mask
#pragma unroll
  for (int j = 0; j < 4; ++j)
#pragma unroll
    for (int vr = 0; vr < 8; ++vr) {
      const int i = (wave << 4) + vr + ((lane >> 4) << 3);
      const int c = (j << 4) + (lane & 15);
      ss[i][c] += sacc[j][vr];
    }
  __syncthreads();

  // softmax: one thread per valid row; normalized p kept in ss + bf16 ps
  if (tid < TOK) {
    const int r = tid;
    float mx = ss[r][0];
    for (int j = 1; j < TOK; ++j) mx = fmaxf(mx, ss[r][j]);
    float sum = 0.f;
    for (int j = 0; j < TOK; ++j) {
      const float e = __expf(ss[r][j] - mx);
      ss[r][j] = e;
      sum += e;
    }
    const float inv = 1.f / sum;
    for (int j = 0; j < TOK; ++j) {
      const float p = ss[r][j] * inv;
      ss[r][j] = p;
      ps[r][j] = f2bf(p);
    }
  }
  __syncthreads();

  // coalesced fp32 attn output from LDS
  {
    float* ao = attn_out + (size_t)bh * (TOK * TOK);
    for (int idx = tid; idx < TOK * TOK; idx += 128) {
      const int i = idx / TOK, j = idx - i * TOK;
      ao[idx] = ss[i][j];
    }
  }

  // out = P @ V : two 16-wide d tiles, K=64 (2 chunks)
#pragma unroll
  for (int j = 0; j < 2; ++j) {
    v8f oacc = vz;
#pragma unroll
    for (int kc = 0; kc < 64; kc += 32) {
      const v16bf a  = frag_a(&ps[(wave << 4) + (lane & 15)][0], lane, kc);
      const v16bf bb = frag_b(&vt[(j << 4) + (lane & 15)][0], lane, kc);
      oacc = wmma_bf16(a, bb, oacc);
    }
#pragma unroll
    for (int vr = 0; vr < 8; ++vr) {
      const int i = (wave << 4) + vr + ((lane >> 4) << 3);
      const int d = (j << 4) + (lane & 15);
      if (i < TOK)
        ctx[((size_t)b * TOK + i) * CH + h * HD + d] = f2bf(oacc[vr]);
    }
  }
}

// ---------------------------------------------------------------------------
extern "C" void kernel_launch(void* const* d_in, const int* in_sizes, int n_in,
                              void* d_out, int out_size, void* d_ws,
                              size_t ws_size, hipStream_t stream) {
  (void)in_sizes; (void)n_in; (void)out_size; (void)ws_size;
  const float* x          = (const float*)d_in[0];
  const float* mask       = (const float*)d_in[1];
  const float* w_qkv      = (const float*)d_in[2];
  const float* b_qkv      = (const float*)d_in[3];
  const float* w_proj     = (const float*)d_in[4];
  const float* b_proj     = (const float*)d_in[5];
  const float* bias_table = (const float*)d_in[6];
  const int*   rel_index  = (const int*)d_in[7];

  const size_t QE = (size_t)NB * NH * TOK * HD;  // 51,380,224
  __bf16* qws = (__bf16*)d_ws;
  __bf16* kws = qws + QE;
  __bf16* vws = kws + QE;
  __bf16* ctx = vws + QE;
  float* bias_hnn = (float*)(ctx + QE);

  float* out  = (float*)d_out;
  float* attn = out + (size_t)NB * TOK * CH;

  bias_gather_kernel<<<(NH * TOK * TOK + 255) / 256, 256, 0, stream>>>(
      bias_table, rel_index, bias_hnn);

  // columns fastest (blockIdx.x) -> A row panel L2-resident across col blocks
  gemm_kernel<0><<<dim3((3 * CH) / 128, ROWS / 128), 256, 0, stream>>>(
      x, w_qkv, b_qkv, qws, kws, vws, nullptr);

  attn_kernel<<<NB * NH, 128, 0, stream>>>(qws, kws, vws, bias_hnn, mask,
                                           attn, ctx);

  gemm_kernel<1><<<dim3(CH / 128, ROWS / 128), 256, 0, stream>>>(
      ctx, w_proj, b_proj, nullptr, nullptr, nullptr, out);
}
